// RecurrentNetworkModel_65575560675507
// MI455X (gfx1250) — matrix operand order repaired and checked by reading, not compiled, not run on hardware
//
#include <hip/hip_runtime.h>
#include <hip/hip_bf16.h>
#include <math.h>

typedef _Float16 f16;
typedef _Float16 v8h  __attribute__((ext_vector_type(8)));
typedef _Float16 v16h __attribute__((ext_vector_type(16)));
typedef float    v8f  __attribute__((ext_vector_type(8)));
typedef unsigned int v4u __attribute__((ext_vector_type(4)));
typedef int      v4i  __attribute__((ext_vector_type(4)));
typedef int      v8i  __attribute__((ext_vector_type(8)));

#define B_   128
#define T_   747
#define K_   64
#define H_   256
#define G3_  768            // 3*H
#define M_   (B_*T_)        // 95616 rows, divisible by 16
#define MT_  (M_/16)        // 5976 M-tiles

// ---- scan kernel LDS layout (bytes) ----
#define HL_STRIDE 264                       // h slice row stride (halfs)
#define ST_STRIDE 772                       // f32 stage row stride (floats)
#define HL_BYTES  (32*HL_STRIDE*2)          // 16896
#define ST_BYTES  (32*ST_STRIDE*4)          // 98816
#define RK_OFF    (HL_BYTES + ST_BYTES)     // 115712
#define RK_ROWS   384                       // rkT columns cached in LDS
#define RK_ROW_STRIDE 264                   // halfs per cached row (512B + 16B TDM pad)
#define RK_CHUNK_BYTES (96*RK_ROW_STRIDE*2) // 50688 per TDM descriptor (96 rows)
#define RK_BYTES  (4*RK_CHUNK_BYTES)        // 202752
#define SMEM_BYTES (RK_OFF + RK_BYTES)      // 318464 <= 320KB WGP LDS

__device__ __forceinline__ v16h cat8(v8h lo, v8h hi) {
  return __builtin_shufflevector(lo, hi, 0,1,2,3,4,5,6,7,8,9,10,11,12,13,14,15);
}
__device__ __forceinline__ v8f wmma16(v16h a, v16h b, v8f c) {
  // D = A(16x32 f16) x B(32x16 f16) + C(16x16 f32)
  return __builtin_amdgcn_wmma_f32_16x16x32_f16(false, a, false, b, (short)0, c, false, false);
}
__device__ __forceinline__ float sigm(float x) { return 1.0f / (1.0f + expf(-x)); }

// ---------------------------------------------------------------- prep kernels

__global__ void cast_f32_f16(const float* __restrict__ src, f16* __restrict__ dst, int n) {
  int i = blockIdx.x * blockDim.x + threadIdx.x;
  int stride = gridDim.x * blockDim.x;
  for (; i < n; i += stride) dst[i] = (f16)src[i];
}

// dst[n][k] = src[k][n] (f32->f16), rows n >= N zero-padded up to Npad
__global__ void transpose_pad_f16(const float* __restrict__ src, f16* __restrict__ dst,
                                  int K, int N, int Npad) {
  int i = blockIdx.x * blockDim.x + threadIdx.x;
  int stride = gridDim.x * blockDim.x;
  int total = K * Npad;
  for (; i < total; i += stride) {
    int n = i / K, k = i - n * K;
    dst[(size_t)n * K + k] = (n < N) ? (f16)src[(size_t)k * N + n] : (f16)0.0f;
  }
}

// ------------------------------------------------ input projection: xg = x @ k + b
// Produces xg in SCAN order: fwd -> xg_f[s][b][:], bwd -> x at time (T-1-s).
__global__ void __launch_bounds__(256) gru_xproj(
    const f16* __restrict__ xh,      // [B][T][64]
    const f16* __restrict__ kT_f,    // [768][64]
    const f16* __restrict__ kT_b,
    const float* __restrict__ bias_f,  // b_fwd row0
    const float* __restrict__ bias_b,
    f16* __restrict__ xg_f,          // [T][B][768]
    f16* __restrict__ xg_b) {
  int wave = threadIdx.x >> 5, lane = threadIdx.x & 31;
  long tile = (long)blockIdx.x * 8 + wave;
  const long tilesPerDir = (long)MT_ * 48;
  int dir = (int)(tile / tilesPerDir);
  long rem = tile - (long)dir * tilesPerDir;
  int mt = (int)(rem / 48), nt = (int)(rem % 48);

  int l15 = lane & 15;
  int kb  = (lane < 16) ? 0 : 8;     // A-matrix K-base split
  int kb2 = (lane < 16) ? 0 : 16;    // unused; B also uses kb
  int mhalf = (lane < 16) ? 0 : 8;   // C-matrix M split

  int mrow = mt * 16 + l15;          // scan-order row: s*128 + b
  int s = mrow >> 7, b = mrow & 127;
  int torig = dir ? (T_ - 1 - s) : s;
  const f16* arow = xh + ((size_t)b * T_ + torig) * K_;
  const f16* kT = dir ? kT_b : kT_f;
  const f16* brow = kT + (size_t)(nt * 16 + l15) * K_;

  v8f acc = {};
  #pragma unroll
  for (int ks = 0; ks < 2; ++ks) {
    int k0 = ks * 32;
    v16h a  = cat8(*(const v8h*)(arow + k0 + kb), *(const v8h*)(arow + k0 + kb + 16));
    v16h bv = cat8(*(const v8h*)(brow + k0 + kb), *(const v8h*)(brow + k0 + kb + 16));
    acc = wmma16(a, bv, acc);
  }

  const float* bias = dir ? bias_b : bias_f;
  f16* xg = dir ? xg_b : xg_f;
  int n = nt * 16 + l15;
  float bn = bias[n];
  #pragma unroll
  for (int v = 0; v < 8; ++v) {
    int m_el = mt * 16 + mhalf + v;
    xg[(size_t)m_el * G3_ + n] = (f16)(acc[v] + bn);
  }
}

// ------------------------------------------------ persistent recurrent scan
// 8 blocks: dir = blk>>2, batch slice of 32 = (blk&3)*32. 1024 threads = 32 waves.
// Per step: inner = h[32,256] @ rkT -> stage (LDS f32), then fused gates update h.
// Columns 0..383 of rkT are TDM-prefetched into LDS (padded rows, conflict-free).
__global__ void __launch_bounds__(1024) gru_scan(
    const f16* __restrict__ xg_f, const f16* __restrict__ xg_b,   // [T][B][768]
    const f16* __restrict__ rkT_f, const f16* __restrict__ rkT_b, // [768][256]
    const float* __restrict__ b_fwd, const float* __restrict__ b_bwd, // [2][768]
    f16* __restrict__ rnn) {                                      // [B][T][512]
  extern __shared__ char smem[];
  f16*   hl = (f16*)smem;                                 // [32][HL_STRIDE]
  float* st = (float*)(smem + HL_BYTES);                  // [32][ST_STRIDE]
  f16*   rkl = (f16*)(smem + RK_OFF);                     // [384][RK_ROW_STRIDE]

  int dir = blockIdx.x >> 2;
  int bslice = (blockIdx.x & 3) * 32;
  const f16* xg  = dir ? xg_b  : xg_f;
  const f16* rkT = dir ? rkT_b : rkT_f;
  const float* rb = (dir ? b_bwd : b_fwd) + G3_;          // recurrent bias row

  int tid = threadIdx.x;
  int lane = tid & 31, wave = tid >> 5;

  // ---- TDM prefill: waves 0..3 each DMA a 48KB strip (96 rkT rows) into LDS,
  // with pad_enable inserting 16B per 512B row -> 528B LDS row stride.
  if (wave < 4) {
    unsigned long long ga = (unsigned long long)(size_t)rkT + (unsigned long long)wave * 49152ull;
    unsigned lds_addr = (unsigned)(size_t)(smem + RK_OFF) + (unsigned)wave * (unsigned)RK_CHUNK_BYTES;
    v4u g0;
    g0[0] = 1u;                                      // count=1 valid descriptor
    g0[1] = lds_addr;                                // lds_addr[31:0]
    g0[2] = (unsigned)ga;                            // global_addr[31:0]
    g0[3] = (unsigned)((ga >> 32) & 0x1FFFFFFu) | (2u << 30);  // addr[56:32] | type=2
    v8i g1;
    g1[0] = (3 << 16)        // data_size = 8B
          | (1 << 20)        // pad_enable
          | (6 << 22)        // pad_interval: 2^6 * 8B = 512B
          | (3 << 25);       // pad_amount: 4 dwords = 16B
    g1[1] = 0x1800 << 16;    // tensor_dim0 = 6144 (low 16 bits at [63:48])
    g1[2] = (1 << 16);       // tensor_dim0 hi = 0; tensor_dim1 = 1 (low bits at [95:80])
    g1[3] = (6144 << 16);    // tensor_dim1 hi = 0; tile_dim0 = 6144 at [127:112]
    g1[4] = 1;               // tile_dim1 = 1, tile_dim2 = 0
    g1[5] = 6144;            // tensor_dim0_stride low 32
    g1[6] = 0;               // stride0 hi = 0; tensor_dim1_stride low = 0 (unused)
    g1[7] = 0;
    v4i g2; g2[0] = 0; g2[1] = 0; g2[2] = 0; g2[3] = 0;
    v4i g3; g3[0] = 0; g3[1] = 0; g3[2] = 0; g3[3] = 0;
    v8i g4; g4[0] = 0; g4[1] = 0; g4[2] = 0; g4[3] = 0;
    g4[4] = 0; g4[5] = 0; g4[6] = 0; g4[7] = 0;
    __builtin_amdgcn_tensor_load_to_lds(g0, g1, g2, g3, g4, 0);
    __builtin_amdgcn_s_wait_tensorcnt(0);
  }
  // zero h while TDM runs
  for (int i = tid; i < 32 * HL_STRIDE; i += 1024) hl[i] = (f16)0.0f;
  __syncthreads();

  int mw = wave >> 4, nw = wave & 15;                     // 2 x 16 wave grid
  int l15 = lane & 15;
  int kb  = (lane < 16) ? 0 : 8;
  int kb2 = (lane < 16) ? 0 : 16;
  int mhalf = (lane < 16) ? 0 : 8;
  const f16* hrow = hl + (size_t)(mw * 16 + l15) * HL_STRIDE;
  const bool useLds = (nw < 8);                           // cols nw*48..+47 < 384
  const f16* bbaseG = rkT + (size_t)(nw * 48 + l15) * H_;
  const f16* bbaseL = rkl + (size_t)(nw * 48 + l15) * RK_ROW_STRIDE;

  // gate-phase ownership: 1024 threads x 8 elems = 32x256
  int gb = tid >> 5;                // local batch row 0..31
  int gj = (tid & 31) * 8;          // 8 consecutive h-columns
  int b  = bslice + gb;

  for (int s = 0; s < T_; ++s) {
    // ---- phase 1: WMMA GEMM into stage
    v8f acc0 = {}, acc1 = {}, acc2 = {};
    #pragma unroll
    for (int ks = 0; ks < 8; ++ks) {
      int k0 = ks * 32;
      v16h a = cat8(*(const v8h*)(hrow + k0 + kb), *(const v8h*)(hrow + k0 + kb + 16));
      v16h b0, b1, b2;
      if (useLds) {
        const f16* bp = bbaseL + k0 + kb;
        b0 = cat8(*(const v8h*)(bp), *(const v8h*)(bp + 16));
        b1 = cat8(*(const v8h*)(bp + 16 * RK_ROW_STRIDE), *(const v8h*)(bp + 16 * RK_ROW_STRIDE + 16));
        b2 = cat8(*(const v8h*)(bp + 32 * RK_ROW_STRIDE), *(const v8h*)(bp + 32 * RK_ROW_STRIDE + 16));
      } else {
        const f16* bp = bbaseG + k0 + kb;
        b0 = cat8(*(const v8h*)(bp), *(const v8h*)(bp + 16));
        b1 = cat8(*(const v8h*)(bp + 16 * H_), *(const v8h*)(bp + 16 * H_ + 16));
        b2 = cat8(*(const v8h*)(bp + 32 * H_), *(const v8h*)(bp + 32 * H_ + 16));
      }
      acc0 = wmma16(a, b0, acc0);
      acc1 = wmma16(a, b1, acc1);
      acc2 = wmma16(a, b2, acc2);
    }
    int srow = mw * 16 + mhalf;
    int scol = nw * 48 + l15;
    #pragma unroll
    for (int v = 0; v < 8; ++v) {
      float* sp = st + (size_t)(srow + v) * ST_STRIDE + scol;
      sp[0] = acc0[v]; sp[16] = acc1[v]; sp[32] = acc2[v];
    }
    __syncthreads();

    // ---- phase 2: gates + h update + output
    const f16* xgr = xg + ((size_t)s * B_ + b) * G3_;
    int tout = dir ? (T_ - 1 - s) : s;
    f16* orow = rnn + ((size_t)b * T_ + tout) * 512 + dir * H_;
    #pragma unroll
    for (int u = 0; u < 8; ++u) {
      int j = gj + u;
      float iz = st[(size_t)gb * ST_STRIDE + j]       + rb[j];
      float ir = st[(size_t)gb * ST_STRIDE + 256 + j] + rb[256 + j];
      float ih = st[(size_t)gb * ST_STRIDE + 512 + j] + rb[512 + j];
      float xz = (float)xgr[j];
      float xr = (float)xgr[256 + j];
      float xh = (float)xgr[512 + j];
      float z  = sigm(xz + iz);
      float r  = sigm(xr + ir);
      float hh = tanhf(xh + r * ih);
      float hold = (float)hl[(size_t)gb * HL_STRIDE + j];
      float hn = z * hold + (1.0f - z) * hh;
      hl[(size_t)gb * HL_STRIDE + j] = (f16)hn;
      orow[j] = (f16)hn;
    }
    __syncthreads();
  }
}

// ------------------------------------------------ MLP head GEMMs
// mode 0: out16[m][Nout] = tanh(A@B + bias)     (h1, h2)
// mode 1: out32[m][nvalid] = A@B + bias, n<nvalid (final logits)
__global__ void __launch_bounds__(256) mlp_gemm(
    const f16* __restrict__ A,    // [M_][Kd]
    const f16* __restrict__ Bt,   // [Nout_pad][Kd]
    const float* __restrict__ bias,
    f16* __restrict__ out16, float* __restrict__ out32,
    int Kd, int Ntiles, int Nout, int mode, int nvalid) {
  int wave = threadIdx.x >> 5, lane = threadIdx.x & 31;
  long tile = (long)blockIdx.x * 8 + wave;
  long total = (long)MT_ * Ntiles;
  if (tile >= total) return;
  int mt = (int)(tile / Ntiles), nt = (int)(tile % Ntiles);

  int l15 = lane & 15;
  int kb  = (lane < 16) ? 0 : 8;
  int kb2 = (lane < 16) ? 0 : 16;
  int mhalf = (lane < 16) ? 0 : 8;
  const f16* arow = A  + (size_t)(mt * 16 + l15) * Kd;
  const f16* brow = Bt + (size_t)(nt * 16 + l15) * Kd;

  v8f acc = {};
  for (int k0 = 0; k0 < Kd; k0 += 32) {
    v16h a  = cat8(*(const v8h*)(arow + k0 + kb), *(const v8h*)(arow + k0 + kb + 16));
    v16h bv = cat8(*(const v8h*)(brow + k0 + kb), *(const v8h*)(brow + k0 + kb + 16));
    acc = wmma16(a, bv, acc);
  }

  int n = nt * 16 + l15;
  bool nok = mode ? (n < nvalid) : true;
  float bn = nok ? bias[n] : 0.0f;
  #pragma unroll
  for (int v = 0; v < 8; ++v) {
    int m_el = mt * 16 + mhalf + v;
    float val = acc[v] + bn;
    if (mode == 0) {
      out16[(size_t)m_el * Nout + n] = (f16)tanhf(val);
    } else if (nok) {
      out32[(size_t)m_el * nvalid + n] = val;
    }
  }
}

// ---------------------------------------------------------------- launcher

extern "C" void kernel_launch(void* const* d_in, const int* in_sizes, int n_in,
                              void* d_out, int out_size, void* d_ws, size_t ws_size,
                              hipStream_t stream) {
  const float* x      = (const float*)d_in[0];
  const float* k_fwd  = (const float*)d_in[1];
  const float* rk_fwd = (const float*)d_in[2];
  const float* b_fwd  = (const float*)d_in[3];
  const float* k_bwd  = (const float*)d_in[4];
  const float* rk_bwd = (const float*)d_in[5];
  const float* b_bwd  = (const float*)d_in[6];
  const float* W1     = (const float*)d_in[7];
  const float* b1     = (const float*)d_in[8];
  const float* W2     = (const float*)d_in[9];
  const float* b2     = (const float*)d_in[10];
  const float* W3     = (const float*)d_in[11];
  const float* b3     = (const float*)d_in[12];
  float* out = (float*)d_out;

  char* ws = (char*)d_ws;
  size_t off = 0;
  auto alloc = [&](size_t bytes) -> char* {
    char* p = ws + off;
    off += (bytes + 255) & ~(size_t)255;
    return p;
  };
  f16* xh    = (f16*)alloc((size_t)B_ * T_ * K_ * 2);
  f16* kT_f  = (f16*)alloc((size_t)G3_ * K_ * 2);
  f16* kT_b  = (f16*)alloc((size_t)G3_ * K_ * 2);
  f16* rkT_f = (f16*)alloc((size_t)G3_ * H_ * 2);
  f16* rkT_b = (f16*)alloc((size_t)G3_ * H_ * 2);
  f16* W1T   = (f16*)alloc((size_t)256 * 512 * 2);
  f16* W2T   = (f16*)alloc((size_t)128 * 256 * 2);
  f16* W3T   = (f16*)alloc((size_t)16 * 128 * 2);
  f16* xg_f  = (f16*)alloc((size_t)T_ * B_ * G3_ * 2);
  f16* xg_b  = (f16*)alloc((size_t)T_ * B_ * G3_ * 2);
  f16* rnn   = (f16*)alloc((size_t)M_ * 512 * 2);
  f16* h1    = (f16*)alloc((size_t)M_ * 256 * 2);
  f16* h2    = (f16*)alloc((size_t)M_ * 128 * 2);

  // ---- prep: casts + weight transposes into [N][K] f16 layouts
  cast_f32_f16<<<4096, 256, 0, stream>>>(x, xh, B_ * T_ * K_);
  transpose_pad_f16<<<256, 256, 0, stream>>>(k_fwd,  kT_f,  K_,  G3_, G3_);
  transpose_pad_f16<<<256, 256, 0, stream>>>(k_bwd,  kT_b,  K_,  G3_, G3_);
  transpose_pad_f16<<<512, 256, 0, stream>>>(rk_fwd, rkT_f, H_,  G3_, G3_);
  transpose_pad_f16<<<512, 256, 0, stream>>>(rk_bwd, rkT_b, H_,  G3_, G3_);
  transpose_pad_f16<<<512, 256, 0, stream>>>(W1,     W1T,   512, 256, 256);
  transpose_pad_f16<<<128, 256, 0, stream>>>(W2,     W2T,   256, 128, 128);
  transpose_pad_f16<<<8,   256, 0, stream>>>(W3,     W3T,   128, 5,   16);

  // ---- input projections (both directions, scan-ordered)
  gru_xproj<<<(2 * MT_ * 48) / 8, 256, 0, stream>>>(xh, kT_f, kT_b, b_fwd, b_bwd, xg_f, xg_b);

  // ---- recurrent scan: 8 persistent workgroups (2 dirs x 4 batch slices)
  (void)hipFuncSetAttribute((const void*)gru_scan,
                            hipFuncAttributeMaxDynamicSharedMemorySize, SMEM_BYTES);
  gru_scan<<<8, 1024, SMEM_BYTES, stream>>>(xg_f, xg_b, rkT_f, rkT_b, b_fwd, b_bwd, rnn);

  // ---- MLP head
  mlp_gemm<<<(MT_ * 16) / 8, 256, 0, stream>>>(rnn, W1T, b1, h1, nullptr, 512, 16, 256, 0, 0);
  mlp_gemm<<<(MT_ * 8)  / 8, 256, 0, stream>>>(h1,  W2T, b2, h2, nullptr, 256, 8,  128, 0, 0);
  mlp_gemm<<<(MT_ * 1 + 7) / 8, 256, 0, stream>>>(h2, W3T, b3, nullptr, out, 128, 1, 16, 1, 5);
}
